// DPM_13391708029991
// MI455X (gfx1250) — matrix operand order repaired and checked by reading, not compiled, not run on hardware
//
#include <hip/hip_runtime.h>

typedef _Float16 half_t;
typedef __attribute__((ext_vector_type(16))) _Float16 v16h;
typedef __attribute__((ext_vector_type(8)))  _Float16 v8h;
typedef __attribute__((ext_vector_type(4)))  _Float16 v4h;
typedef __attribute__((ext_vector_type(8)))  float    v8f;

#define TB   256
#define BM   64
#define BK   32
#define ASTR 40      // LDS row stride in halfs (80 B, 16-B aligned rows)
#define MAXK 6272    // largest K (conv4: 128*7*7)

// ---------------------------------------------------------------------------
// Implicit-GEMM conv with WMMA f16 (f32 accumulate), fused scale/bias (+ReLU).
//   in : (B, Cin, IH, IW) f16, NCHW, VALID conv
//   wT : (K, Cout) f16, K = Cin*KH*KW (pre-transposed; +256B slack allocated)
// Block: 256 thr = 8 waves; tile BM(64) x BNT(64|128); wave = 16M x (BNT/2)N.
// Double-buffered LDS, one barrier per K-step; unconditional clamped loads.
// All B fragments are loaded before the WMMA chain so partial s_wait_dscnt
// overlaps LDS latency with matrix issue.
// ---------------------------------------------------------------------------
template <typename OutT, bool RELU, int BNT>
__global__ __launch_bounds__(TB) void wmma_conv(
    const half_t* __restrict__ in, const half_t* __restrict__ wT,
    const float* __restrict__ alpha, const float* __restrict__ beta,
    OutT* __restrict__ out,
    int Bn, int Cin, int IH, int IW, int Cout, int KH, int KW,
    int OH, int OW, long outChStride, long outBatchStride)
{
    constexpr int NA = BNT / 32;   // WMMA accumulators per wave; also v4h B loads/thr

    __shared__ __align__(16) half_t Alds[2][BM][ASTR];
    __shared__ __align__(16) half_t Blds[2][BNT][ASTR];   // transposed: [n][k]
    __shared__ int aoff[MAXK];

    const int K    = Cin * KH * KW;
    const int HWo  = OH * OW;
    const int Mtot = Bn * HWo;
    const int tid  = threadIdx.x;
    const int lane = tid & 31;
    const int wid  = tid >> 5;
    const int wm   = (wid & 3) << 4;          // wave M offset in tile
    const int wn   = (wid >> 2) * (BNT / 2);  // wave N offset in tile
    const int mBase = blockIdx.x * BM;
    const int nBase = blockIdx.y * BNT;

    // im2col offset table, built once per block (no divisions in hot loop)
    const int khw = KH * KW;
    for (int k = tid; k < K; k += TB) {
        int ci = k / khw;
        int r  = k - ci * khw;
        int ky = r / KW;
        int kx = r - ky * KW;
        aoff[k] = (ci * IH + ky) * IW + kx;
    }

    // A staging: this thread fills column `lane` of rows wid, wid+8, ..., wid+56.
    // Clamp invalid rows to pixel 0: loads stay in-bounds; garbage lands only in
    // D rows >= Mtot which are never stored.
    long p0_[8];
#pragma unroll
    for (int j = 0; j < 8; ++j) {
        int m = mBase + wid + 8 * j;
        int mc = (m < Mtot) ? m : 0;
        int b  = mc / HWo;
        int pix = mc - b * HWo;
        int oy = pix / OW;
        int ox = pix - oy * OW;
        p0_[j] = ((long)b * Cin * IH + oy) * (long)IW + ox;
    }
    __syncthreads();   // aoff table ready

    half_t av[8];
    v4h    bv[NA];

    auto gfetch = [&](int k0) {
        int off = aoff[k0 + lane];            // single LDS read per step
#pragma unroll
        for (int j = 0; j < 8; ++j) av[j] = in[p0_[j] + off];
#pragma unroll
        for (int jj = 0; jj < NA; ++jj) {
            int linear = (tid + TB * jj) * 4; // 4 consecutive chans per load
            int ni = linear % BNT;
            int kk = linear / BNT;
            bv[jj] = *(const v4h*)&wT[(long)(k0 + kk) * Cout + nBase + ni];
        }
    };
    auto lput = [&](int p) {
#pragma unroll
        for (int j = 0; j < 8; ++j) Alds[p][wid + 8 * j][lane] = av[j];
#pragma unroll
        for (int jj = 0; jj < NA; ++jj) {
            int linear = (tid + TB * jj) * 4;
            int ni = linear % BNT;
            int kk = linear / BNT;
#pragma unroll
            for (int e = 0; e < 4; ++e) Blds[p][ni + e][kk] = bv[jj][e];
        }
    };

    v8f c[NA] = {};

    gfetch(0);
    lput(0);
    __syncthreads();

    int p = 0;
    for (int k0 = 0; k0 < K; k0 += BK) {      // K is a multiple of 32 everywhere
        const bool more = (k0 + BK) < K;
        if (more) gfetch(k0 + BK);            // overlap global latency with WMMA

        // ---- fragments per documented CDNA5 WMMA VGPR layouts ----
        const int arow = wm + (lane & 15);
        const int kb   = (lane >> 4) << 3;    // A: 0 or 8
        const int kb2  = (lane >> 4) << 4;    // unused; B loads use kb, same k order as A
        v8h alo = *(const v8h*)&Alds[p][arow][kb];
        v8h ahi = *(const v8h*)&Alds[p][arow][kb + 16];
        v16h a;
#pragma unroll
        for (int i = 0; i < 8; ++i) { a[i] = alo[i]; a[i + 8] = ahi[i]; }

        // load ALL B fragments first, then issue the WMMA chain: allows partial
        // s_wait_dscnt per WMMA instead of a full LDS-queue drain each time
        const int bcol = lane & 15;
        v16h bfr[NA];
#pragma unroll
        for (int h = 0; h < NA; ++h) {
            v8h blo = *(const v8h*)&Blds[p][wn + h * 16 + bcol][kb];
            v8h bhi = *(const v8h*)&Blds[p][wn + h * 16 + bcol][kb + 16];
#pragma unroll
            for (int i = 0; i < 8; ++i) { bfr[h][i] = blo[i]; bfr[h][i + 8] = bhi[i]; }
        }
#pragma unroll
        for (int h = 0; h < NA; ++h) {
            c[h] = __builtin_amdgcn_wmma_f32_16x16x32_f16(false, a, false, bfr[h],
                                                          (short)0, c[h],
                                                          false, false);
        }

        if (more) lput(p ^ 1);
        __syncthreads();                      // one barrier per K-step
        p ^= 1;
    }

    // ---- epilogue: D layout = VGPR r -> row r + 8*(lane>=16), col = lane&15 ----
    const int rowoff = (lane >> 4) << 3;
    const int col    = lane & 15;
#pragma unroll
    for (int r = 0; r < 8; ++r) {
        int m = mBase + wm + rowoff + r;
        if (m >= Mtot) continue;
        int b   = m / HWo;
        int pix = m - b * HWo;
#pragma unroll
        for (int h = 0; h < NA; ++h) {
            int n = nBase + wn + (h << 4) + col;
            if (n >= Cout) continue;
            float y = c[h][r] * alpha[n] + beta[n];
            if (RELU) y = fmaxf(y, 0.0f);
            out[(long)b * outBatchStride + (long)n * outChStride + pix] = (OutT)y;
        }
    }
}

// ---------------------------------------------------------------------------
// conv1: reflect-pad(7) + 3x3 VALID, 3->64, BN+ReLU fused, f16 out. K=27 (tiny).
// ---------------------------------------------------------------------------
__global__ __launch_bounds__(256) void conv1_reflect(
    const float* __restrict__ src, const float* __restrict__ w,
    const float* __restrict__ alpha, const float* __restrict__ beta,
    half_t* __restrict__ out, int Bn, int H, int W)
{
    const int OH = H + 12, OW = W + 12;
    long idx = (long)blockIdx.x * blockDim.x + threadIdx.x;
    long tot = (long)Bn * OH * OW;
    if (idx >= tot) return;
    int b   = (int)(idx / (OH * OW));
    int pix = (int)(idx - (long)b * OH * OW);
    int oy = pix / OW, ox = pix - oy * OW;

    float patch[27];
#pragma unroll
    for (int ci = 0; ci < 3; ++ci)
#pragma unroll
        for (int ky = 0; ky < 3; ++ky)
#pragma unroll
            for (int kx = 0; kx < 3; ++kx) {
                int qy = oy + ky - 7;
                int qx = ox + kx - 7;
                qy = qy < 0 ? -qy : qy; if (qy >= H) qy = 2 * H - 2 - qy;
                qx = qx < 0 ? -qx : qx; if (qx >= W) qx = 2 * W - 2 - qx;
                patch[(ci * 3 + ky) * 3 + kx] =
                    src[((long)b * 3 + ci) * H * W + (long)qy * W + qx];
            }
    long outPix = (long)oy * OW + ox;
    for (int co = 0; co < 64; ++co) {
        float acc = 0.0f;
#pragma unroll
        for (int t = 0; t < 27; ++t) acc += patch[t] * w[co * 27 + t];
        float y = fmaxf(acc * alpha[co] + beta[co], 0.0f);
        out[((long)b * 64 + co) * OH * OW + outPix] = (half_t)y;
    }
}

// ---------------------------------------------------------------------------
// ZM polar conv: two 13x13 convs (3->16, zero pad 6), magnitude. f32 out.
// ---------------------------------------------------------------------------
__global__ __launch_bounds__(256) void zm_conv(
    const float* __restrict__ src,
    const float* __restrict__ wr, const float* __restrict__ wi,
    const float* __restrict__ rb, const float* __restrict__ ib,
    float* __restrict__ out, int Bn, int H, int W, long outBatchStride)
{
    long idx = (long)blockIdx.x * blockDim.x + threadIdx.x;
    long tot = (long)Bn * 16 * H * W;
    if (idx >= tot) return;
    int b   = (int)(idx / (16L * H * W));
    long r0 = idx - (long)b * 16 * H * W;
    int co  = (int)(r0 / (H * W));
    int pix = (int)(r0 - (long)co * H * W);
    int oy = pix / W, ox = pix - oy * W;

    float ar = rb[co], ai = ib[co];
    for (int ci = 0; ci < 3; ++ci) {
        const float* sp  = src + ((long)b * 3 + ci) * H * W;
        const float* wrp = wr + ((long)co * 3 + ci) * 169;
        const float* wip = wi + ((long)co * 3 + ci) * 169;
        for (int ky = 0; ky < 13; ++ky) {
            int iy = oy + ky - 6;
            if (iy < 0 || iy >= H) continue;
#pragma unroll
            for (int kx = 0; kx < 13; ++kx) {
                int ix = ox + kx - 6;
                if (ix < 0 || ix >= W) continue;
                float xv = sp[(long)iy * W + ix];
                ar += xv * wrp[ky * 13 + kx];
                ai += xv * wip[ky * 13 + kx];
            }
        }
    }
    out[(long)b * outBatchStride + (long)co * H * W + (long)oy * W + ox] =
        sqrtf(ar * ar + ai * ai + 1e-10f);
}

// ---------------------------------------------------------------------------
// Bilinear resize, align_corners=True. Strided src/dst (dst may be a channel
// slice of the concatenated output tensor).
// ---------------------------------------------------------------------------
__global__ __launch_bounds__(256) void resize_ac(
    const float* __restrict__ src, float* __restrict__ dst,
    int Bn, int C, int SH, int SW, int OH, int OW,
    long srcBatchStride, long dstBatchStride)
{
    long idx = (long)blockIdx.x * blockDim.x + threadIdx.x;
    long tot = (long)Bn * C * OH * OW;
    if (idx >= tot) return;
    int b   = (int)(idx / ((long)C * OH * OW));
    long r0 = idx - (long)b * C * OH * OW;
    int c   = (int)(r0 / (OH * OW));
    int pix = (int)(r0 - (long)c * OH * OW);
    int oy = pix / OW, ox = pix - oy * OW;

    float sy = (OH > 1) ? (float)(SH - 1) / (float)(OH - 1) : 0.0f;
    float sx = (OW > 1) ? (float)(SW - 1) / (float)(OW - 1) : 0.0f;
    float ys = oy * sy, xs = ox * sx;
    int y0 = (int)floorf(ys); int y1 = (y0 + 1 < SH) ? y0 + 1 : SH - 1;
    int x0 = (int)floorf(xs); int x1 = (x0 + 1 < SW) ? x0 + 1 : SW - 1;
    float wy = ys - (float)y0, wx = xs - (float)x0;

    const float* sp = src + (long)b * srcBatchStride + (long)c * SH * SW;
    float v00 = sp[(long)y0 * SW + x0], v01 = sp[(long)y0 * SW + x1];
    float v10 = sp[(long)y1 * SW + x0], v11 = sp[(long)y1 * SW + x1];
    float top = v00 + (v01 - v00) * wx;
    float bot = v10 + (v11 - v10) * wx;
    dst[(long)b * dstBatchStride + (long)c * OH * OW + (long)oy * OW + ox] =
        top + (bot - top) * wy;
}

// ---------------------------------------------------------------------------
// Weight prep: OIHW f32 -> (K, Cout) f16 transposed (coalesced writes).
// ---------------------------------------------------------------------------
__global__ __launch_bounds__(256) void wtrans(
    const float* __restrict__ w, half_t* __restrict__ wT, int Cout, int K)
{
    long idx = (long)blockIdx.x * blockDim.x + threadIdx.x;
    if (idx >= (long)Cout * K) return;
    int co = (int)(idx % Cout);
    int k  = (int)(idx / Cout);
    wT[idx] = (half_t)w[(long)co * K + k];
}

// Fold BN (running mean=0, var=1) + conv bias into per-channel alpha/beta.
__global__ void bnprep(
    const float* b1, const float* g1, const float* be1, float* a1, float* bb1,
    const float* b2, const float* g2, const float* be2, float* a2, float* bb2,
    const float* b3, const float* g3, const float* be3, float* a3, float* bb3,
    const float* b4, const float* g4, const float* be4, float* a4, float* bb4,
    const float* b5, float* a5, float* bb5)
{
    int i = threadIdx.x;
    const float rs = rsqrtf(1.0f + 1e-5f);
    if (i < 64)  { float s = g1[i] * rs; a1[i] = s; bb1[i] = b1[i] * s + be1[i]; }
    if (i < 128) { float s = g2[i] * rs; a2[i] = s; bb2[i] = b2[i] * s + be2[i]; }
    if (i < 128) { float s = g3[i] * rs; a3[i] = s; bb3[i] = b3[i] * s + be3[i]; }
    if (i < 64)  { float s = g4[i] * rs; a4[i] = s; bb4[i] = b4[i] * s + be4[i]; }
    if (i < 32)  { a5[i] = 1.0f; bb5[i] = b5[i]; }
}

// ---------------------------------------------------------------------------
extern "C" void kernel_launch(void* const* d_in, const int* in_sizes, int n_in,
                              void* d_out, int out_size, void* d_ws, size_t ws_size,
                              hipStream_t stream)
{
    (void)in_sizes; (void)n_in; (void)out_size; (void)ws_size;

    const float* x    = (const float*)d_in[0];
    const float* zmrw = (const float*)d_in[1];
    const float* zmiw = (const float*)d_in[2];
    const float* zmrb = (const float*)d_in[3];
    const float* zmib = (const float*)d_in[4];
    const float* w1 = (const float*)d_in[5];
    const float* b1 = (const float*)d_in[6];
    const float* g1 = (const float*)d_in[7];
    const float* be1= (const float*)d_in[8];
    const float* w2 = (const float*)d_in[9];
    const float* b2 = (const float*)d_in[10];
    const float* g2 = (const float*)d_in[11];
    const float* be2= (const float*)d_in[12];
    const float* w3 = (const float*)d_in[13];
    const float* b3 = (const float*)d_in[14];
    const float* g3 = (const float*)d_in[15];
    const float* be3= (const float*)d_in[16];
    const float* w4 = (const float*)d_in[17];
    const float* b4 = (const float*)d_in[18];
    const float* g4 = (const float*)d_in[19];
    const float* be4= (const float*)d_in[20];
    const float* w5 = (const float*)d_in[21];
    const float* b5 = (const float*)d_in[22];

    const int B = 4, H0 = 256, W0 = 256;
    const int Hs = 170, Ws = 170;   // int(256/1.5)
    const int Hl = 341, Wl = 341;   // int(256/0.75)

    char* ws = (char*)d_ws;
    size_t off = 0;
    auto alloc = [&](size_t bytes) -> char* {
        char* p = ws + off;
        off = (off + bytes + 255) & ~(size_t)255;
        return p;
    };
    float*  sd   = (float*)alloc((size_t)B * 3 * Hs * Ws * 4);
    float*  su   = (float*)alloc((size_t)B * 3 * Hl * Wl * 4);
    half_t* wT2  = (half_t*)alloc((size_t)576 * 128 * 2 + 256);
    half_t* wT3  = (half_t*)alloc((size_t)3200 * 128 * 2 + 256);
    half_t* wT4  = (half_t*)alloc((size_t)6272 * 64 * 2 + 256);
    half_t* wT5  = (half_t*)alloc((size_t)64 * 32 * 2 + 256);
    float*  a1 = (float*)alloc(64 * 4);   float* bb1 = (float*)alloc(64 * 4);
    float*  a2 = (float*)alloc(128 * 4);  float* bb2 = (float*)alloc(128 * 4);
    float*  a3 = (float*)alloc(128 * 4);  float* bb3 = (float*)alloc(128 * 4);
    float*  a4 = (float*)alloc(64 * 4);   float* bb4 = (float*)alloc(64 * 4);
    float*  a5 = (float*)alloc(32 * 4);   float* bb5 = (float*)alloc(32 * 4);
    half_t* t1   = (half_t*)alloc((size_t)B * 64  * 353 * 353 * 2);
    half_t* t2   = (half_t*)alloc((size_t)B * 128 * 351 * 351 * 2);
    half_t* t3   = (half_t*)alloc((size_t)B * 128 * 347 * 347 * 2);
    half_t* t4   = (half_t*)alloc((size_t)B * 64  * 341 * 341 * 2);
    float*  headf= (float*)alloc((size_t)B * 32 * 341 * 341 * 4);
    float*  zmf  = (float*)alloc((size_t)B * 16 * 341 * 341 * 4);

    auto cdiv = [](long a, long b) { return (unsigned)((a + b - 1) / b); };

    wtrans<<<cdiv((long)128 * 576, 256), 256, 0, stream>>>(w2, wT2, 128, 576);
    wtrans<<<cdiv((long)128 * 3200, 256), 256, 0, stream>>>(w3, wT3, 128, 3200);
    wtrans<<<cdiv((long)64 * 6272, 256), 256, 0, stream>>>(w4, wT4, 64, 6272);
    wtrans<<<cdiv((long)32 * 64, 256), 256, 0, stream>>>(w5, wT5, 32, 64);
    bnprep<<<1, 128, 0, stream>>>(b1, g1, be1, a1, bb1,
                                  b2, g2, be2, a2, bb2,
                                  b3, g3, be3, a3, bb3,
                                  b4, g4, be4, a4, bb4,
                                  b5, a5, bb5);

    resize_ac<<<cdiv((long)B * 3 * Hs * Ws, 256), 256, 0, stream>>>(
        x, sd, B, 3, H0, W0, Hs, Ws, (long)3 * H0 * W0, (long)3 * Hs * Ws);
    resize_ac<<<cdiv((long)B * 3 * Hl * Wl, 256), 256, 0, stream>>>(
        x, su, B, 3, H0, W0, Hl, Wl, (long)3 * H0 * W0, (long)3 * Hl * Wl);

    float* out_f = (float*)d_out;
    const long OB = 144L * H0 * W0;

    auto run_scale = [&](const float* src, int H, int W,
                         float* zm_out, long zmBatch,
                         float* head_out, long headBatch) {
        zm_conv<<<cdiv((long)B * 16 * H * W, 256), 256, 0, stream>>>(
            src, zmrw, zmiw, zmrb, zmib, zm_out, B, H, W, zmBatch);
        int H1 = H + 12, W1 = W + 12;
        conv1_reflect<<<cdiv((long)B * H1 * W1, 256), 256, 0, stream>>>(
            src, w1, a1, bb1, t1, B, H, W);
        {
            int OHc = H + 10, OWc = W + 10;
            long M = (long)B * OHc * OWc;
            wmma_conv<half_t, true, 128><<<dim3(cdiv(M, BM)), TB, 0, stream>>>(
                t1, wT2, a2, bb2, t2, B, 64, H1, W1, 128, 3, 3,
                OHc, OWc, (long)OHc * OWc, 128L * OHc * OWc);
        }
        {
            int OHc = H + 6, OWc = W + 6;
            long M = (long)B * OHc * OWc;
            wmma_conv<half_t, true, 128><<<dim3(cdiv(M, BM)), TB, 0, stream>>>(
                t2, wT3, a3, bb3, t3, B, 128, H + 10, W + 10, 128, 5, 5,
                OHc, OWc, (long)OHc * OWc, 128L * OHc * OWc);
        }
        {
            long M = (long)B * H * W;
            wmma_conv<half_t, true, 64><<<dim3(cdiv(M, BM)), TB, 0, stream>>>(
                t3, wT4, a4, bb4, t4, B, 128, H + 6, W + 6, 64, 7, 7,
                H, W, (long)H * W, 64L * H * W);
        }
        {
            long M = (long)B * H * W;
            wmma_conv<float, false, 64><<<dim3(cdiv(M, BM)), TB, 0, stream>>>(
                t4, wT5, a5, bb5, head_out, B, 64, H, W, 32, 1, 1,
                H, W, (long)H * W, headBatch);
        }
    };

    run_scale(su, Hl, Wl, zmf, 16L * Hl * Wl, headf, 32L * Hl * Wl);
    resize_ac<<<cdiv((long)B * 16 * H0 * W0, 256), 256, 0, stream>>>(
        zmf, out_f + 0L * H0 * W0, B, 16, Hl, Wl, H0, W0, 16L * Hl * Wl, OB);
    resize_ac<<<cdiv((long)B * 32 * H0 * W0, 256), 256, 0, stream>>>(
        headf, out_f + 48L * H0 * W0, B, 32, Hl, Wl, H0, W0, 32L * Hl * Wl, OB);

    run_scale(x, H0, W0, out_f + 16L * H0 * W0, OB, out_f + 80L * H0 * W0, OB);

    run_scale(sd, Hs, Ws, zmf, 16L * Hs * Ws, headf, 32L * Hs * Ws);
    resize_ac<<<cdiv((long)B * 16 * H0 * W0, 256), 256, 0, stream>>>(
        zmf, out_f + 32L * H0 * W0, B, 16, Hs, Ws, H0, W0, 16L * Hs * Ws, OB);
    resize_ac<<<cdiv((long)B * 32 * H0 * W0, 256), 256, 0, stream>>>(
        headf, out_f + 112L * H0 * W0, B, 32, Hs, Ws, H0, W0, 32L * Hs * Ws, OB);
}
